// CausalAttention_90950227460101
// MI455X (gfx1250) — matrix operand order repaired and checked by reading, not compiled, not run on hardware
//
#include <hip/hip_runtime.h>

#define T_SEQ   2048
#define D_MODEL 2048
#define KV_DIM  512
#define HEAD_D  64
#define N_HEADS 32
#define N_KVH   8

typedef __attribute__((ext_vector_type(4)))  unsigned int uint4v;
typedef __attribute__((ext_vector_type(8)))  unsigned int uint8v;
typedef __attribute__((ext_vector_type(16))) __bf16       bf16x16;
typedef __attribute__((ext_vector_type(8)))  float        f32x8;

union Frag {
  struct { uint4v lo, hi; } q;
  uint8v  u;
  bf16x16 bf;
};

__device__ __forceinline__ unsigned short f2bf(float f) {
  unsigned int u = __builtin_bit_cast(unsigned int, f);
  unsigned int r = u + 0x7FFFu + ((u >> 16) & 1u);   // round-to-nearest-even
  return (unsigned short)(r >> 16);
}

__device__ __forceinline__ f32x8 wmma_bf16(const Frag& a, const Frag& b, f32x8 c) {
  return __builtin_amdgcn_wmma_f32_16x16x32_bf16(false, a.bf, false, b.bf,
                                                 (short)0, c, false, false);
}

__device__ __forceinline__ void ld_frag(Frag& f, const unsigned short* p, int hiOff) {
  f.q.lo = *(const uint4v*)(p);
  f.q.hi = *(const uint4v*)(p + hiOff);
}

// ---------------------------------------------------------------- casts
__global__ void cast_f32_bf16_kernel(const float* __restrict__ s,
                                     unsigned short* __restrict__ d, int n) {
  int i = blockIdx.x * blockDim.x + threadIdx.x;
  int stride = gridDim.x * blockDim.x;
  for (; i < n; i += stride) d[i] = f2bf(s[i]);
}

// ---------------------------------------------------------------- RoPE tables (llama3 scaling)
__global__ void rope_tables_kernel(float* __restrict__ cosT, float* __restrict__ sinT) {
  int idx = blockIdx.x * blockDim.x + threadIdx.x;       // t*32 + i
  if (idx >= T_SEQ * 32) return;
  int t = idx >> 5, i = idx & 31;
  float inv = __powf(500000.0f, -(2.0f * (float)i) / 64.0f);
  float wavelen = 6.28318530718f / inv;
  float low_wl = 8192.0f, high_wl = 2048.0f;             // orig_ctx / {low,high}_freq_factor
  float inv_l = (wavelen > low_wl) ? inv / 32.0f : inv;
  float smooth = (8192.0f / wavelen - 1.0f) / 3.0f;
  float smoothed = (1.0f - smooth) * (inv / 32.0f) + smooth * inv;
  float invf = (wavelen <= low_wl && wavelen >= high_wl) ? smoothed : inv_l;
  float ang = (float)t * invf;
  cosT[idx] = __cosf(ang);
  sinT[idx] = __sinf(ang);
}

// ---------------------------------------------------------------- RoPE apply (f32 -> bf16), scale folds 1/sqrt(HD) for Q
__global__ void rope_apply_kernel(const float* __restrict__ X, unsigned short* __restrict__ Y,
                                  const float* __restrict__ cosT, const float* __restrict__ sinT,
                                  int nHeads, float scale) {
  int idx = blockIdx.x * blockDim.x + threadIdx.x;
  if (idx >= T_SEQ * nHeads * 32) return;
  int i = idx & 31;
  int h = (idx >> 5) % nHeads;
  int t = idx / (32 * nHeads);
  size_t base = (size_t)t * nHeads * HEAD_D + (size_t)h * HEAD_D;
  float x1 = X[base + i], x2 = X[base + 32 + i];
  float c = cosT[t * 32 + i], s = sinT[t * 32 + i];
  Y[base + i]      = f2bf((x1 * c - x2 * s) * scale);
  Y[base + 32 + i] = f2bf((x2 * c + x1 * s) * scale);
}

// ---------------------------------------------------------------- V -> Vt bf16  [kvh*64+f][token]
__global__ void transpose_v_kernel(const float* __restrict__ Vf, unsigned short* __restrict__ Vt) {
  int idx = blockIdx.x * blockDim.x + threadIdx.x;
  if (idx >= T_SEQ * KV_DIM) return;
  int t = idx / KV_DIM, c = idx % KV_DIM;
  Vt[(size_t)c * T_SEQ + t] = f2bf(Vf[(size_t)t * KV_DIM + c]);
}

// ---------------------------------------------------------------- C = A(MxK) * B(NxK)^T, bf16 in / f32 out
// one wave computes a 16x64 output tile; k-loop is software-pipelined
// (next k-step's fragments are in flight while current WMMAs execute)
__global__ void gemm_bf16_nt_kernel(const unsigned short* __restrict__ A,
                                    const unsigned short* __restrict__ B,
                                    float* __restrict__ C, int M, int N, int K) {
  const int lane = threadIdx.x & 31;
  const int wave = threadIdx.x >> 5;
  const int half = lane >> 4;
  const int l16  = lane & 15;
  const int tilesN = N >> 6;
  const int tile = blockIdx.x * (blockDim.x >> 5) + wave;
  if (tile >= (M >> 4) * tilesN) return;
  const int row0 = (tile / tilesN) << 4;
  const int col0 = (tile % tilesN) << 6;

  f32x8 acc0 = {}, acc1 = {}, acc2 = {}, acc3 = {};
  const unsigned short* arow  = A + (size_t)(row0 + l16) * K + 16 * half;
  const unsigned short* brow0 = B + (size_t)(col0 + l16) * K + 16 * half;
  const unsigned short* brow1 = brow0 + (size_t)16 * K;
  const unsigned short* brow2 = brow0 + (size_t)32 * K;
  const unsigned short* brow3 = brow0 + (size_t)48 * K;

  Frag a, b0, b1, b2, b3;
  ld_frag(a,  arow,  8);
  ld_frag(b0, brow0, 8);
  ld_frag(b1, brow1, 8);
  ld_frag(b2, brow2, 8);
  ld_frag(b3, brow3, 8);

  for (int k0 = 32; k0 < K; k0 += 32) {
    Frag an, b0n, b1n, b2n, b3n;
    ld_frag(an,  arow  + k0, 8);
    ld_frag(b0n, brow0 + k0, 8);
    ld_frag(b1n, brow1 + k0, 8);
    ld_frag(b2n, brow2 + k0, 8);
    ld_frag(b3n, brow3 + k0, 8);
    acc0 = wmma_bf16(a, b0, acc0);
    acc1 = wmma_bf16(a, b1, acc1);
    acc2 = wmma_bf16(a, b2, acc2);
    acc3 = wmma_bf16(a, b3, acc3);
    a = an; b0 = b0n; b1 = b1n; b2 = b2n; b3 = b3n;
  }
  acc0 = wmma_bf16(a, b0, acc0);
  acc1 = wmma_bf16(a, b1, acc1);
  acc2 = wmma_bf16(a, b2, acc2);
  acc3 = wmma_bf16(a, b3, acc3);

  float* crow = C + (size_t)(row0 + 8 * half) * N + col0 + l16;
  for (int r = 0; r < 8; ++r) {
    crow[0]  = acc0[r];
    crow[16] = acc1[r];
    crow[32] = acc2[r];
    crow[48] = acc3[r];
    crow += N;
  }
}

// ---------------------------------------------------------------- flash attention, 1 wave = 16 queries of one head
// K-fragments for the NEXT kv block and V-fragments for THIS block are issued
// right after the S=QK^T WMMAs so their latency hides under the softmax VALU work.
__global__ void attention_kernel(const unsigned short* __restrict__ Qb,   // [T][2048] bf16 (pre-scaled by 1/8)
                                 const unsigned short* __restrict__ Kb,   // [T][512]  bf16 (roped)
                                 const unsigned short* __restrict__ Vt,   // [512][T]  bf16 (transposed)
                                 unsigned short* __restrict__ Ctxb) {     // [T][2048] bf16
  __shared__ __align__(16) unsigned short Plds[4][16 * 32];
  const int lane = threadIdx.x & 31;
  const int wave = threadIdx.x >> 5;
  const int half = lane >> 4;
  const int l16  = lane & 15;
  const int head   = blockIdx.x >> 5;        // 32 q-chunks per head
  const int qchunk = blockIdx.x & 31;
  const int q0  = qchunk * 64 + wave * 16;
  const int kvh = head >> 2;                 // 4 Q heads per KV head
  const int kvEnd = q0 + 16;

  const unsigned short* qrow = Qb + (size_t)(q0 + l16) * D_MODEL + head * HEAD_D + 16 * half;
  Frag aq0, aq1;
  ld_frag(aq0, qrow,      8);
  ld_frag(aq1, qrow + 32, 8);

  f32x8 o0 = {}, o1 = {}, o2 = {}, o3 = {};
  float mrow[8], lrow[8];
  for (int r = 0; r < 8; ++r) { mrow[r] = -__builtin_inff(); lrow[r] = 0.0f; }

  unsigned short* pw = &Plds[wave][0];
  const unsigned short* krow = Kb + (size_t)l16 * KV_DIM + kvh * HEAD_D + 16 * half;
  const unsigned short* vrow = Vt + (size_t)(kvh * HEAD_D + l16) * T_SEQ + 16 * half;

  // preload K-fragments for kv0 = 0
  Frag bk00, bk01, bk10, bk11;
  ld_frag(bk00, krow, 8);
  ld_frag(bk01, krow + 32, 8);
  ld_frag(bk10, krow + (size_t)16 * KV_DIM, 8);
  ld_frag(bk11, krow + (size_t)16 * KV_DIM + 32, 8);

  for (int kv0 = 0; kv0 < kvEnd; kv0 += 32) {
    f32x8 s0 = {}, s1 = {};
    s0 = wmma_bf16(aq0, bk00, s0);
    s0 = wmma_bf16(aq1, bk01, s0);
    s1 = wmma_bf16(aq0, bk10, s1);
    s1 = wmma_bf16(aq1, bk11, s1);

    // issue next KV block's K-fragments now (hidden under softmax below)
    const int kvn = kv0 + 32;
    if (kvn < kvEnd) {
      const unsigned short* krn = krow + (size_t)kvn * KV_DIM;
      ld_frag(bk00, krn, 8);
      ld_frag(bk01, krn + 32, 8);
      ld_frag(bk10, krn + (size_t)16 * KV_DIM, 8);
      ld_frag(bk11, krn + (size_t)16 * KV_DIM + 32, 8);
    }
    // issue this block's V-fragments now as well
    Frag bv0, bv1, bv2, bv3;
    ld_frag(bv0, vrow + kv0, 8);
    ld_frag(bv1, vrow + kv0 + (size_t)16 * T_SEQ, 8);
    ld_frag(bv2, vrow + kv0 + (size_t)32 * T_SEQ, 8);
    ld_frag(bv3, vrow + kv0 + (size_t)48 * T_SEQ, 8);

    const int c0 = kv0 + l16;
    const int c1 = kv0 + 16 + l16;
    for (int r = 0; r < 8; ++r) {
      const int row = q0 + 8 * half + r;
      float v0 = (c0 <= row) ? s0[r] : -__builtin_inff();
      float v1 = (c1 <= row) ? s1[r] : -__builtin_inff();
      float t = fmaxf(v0, v1);
      t = fmaxf(t, __shfl_xor(t, 1, 32));
      t = fmaxf(t, __shfl_xor(t, 2, 32));
      t = fmaxf(t, __shfl_xor(t, 4, 32));
      t = fmaxf(t, __shfl_xor(t, 8, 32));
      const float mnew = fmaxf(mrow[r], t);
      const float sc = __expf(mrow[r] - mnew);
      const float p0 = __expf(v0 - mnew);
      const float p1 = __expf(v1 - mnew);
      float ps = p0 + p1;
      ps += __shfl_xor(ps, 1, 32);
      ps += __shfl_xor(ps, 2, 32);
      ps += __shfl_xor(ps, 4, 32);
      ps += __shfl_xor(ps, 8, 32);
      lrow[r] = lrow[r] * sc + ps;
      mrow[r] = mnew;
      o0[r] *= sc; o1[r] *= sc; o2[r] *= sc; o3[r] *= sc;
      pw[(8 * half + r) * 32 + l16]      = f2bf(p0);   // P row-major 16x32 in LDS
      pw[(8 * half + r) * 32 + 16 + l16] = f2bf(p1);
    }
    asm volatile("s_wait_dscnt 0" ::: "memory");       // wave-private LDS RAW

    // re-load P as an A-frag (16x32) from LDS
    Frag ap;
    ld_frag(ap, pw + l16 * 32 + 16 * half, 8);

    o0 = wmma_bf16(ap, bv0, o0);
    o1 = wmma_bf16(ap, bv1, o1);
    o2 = wmma_bf16(ap, bv2, o2);
    o3 = wmma_bf16(ap, bv3, o3);
  }

  for (int r = 0; r < 8; ++r) {
    const float inv = 1.0f / lrow[r];
    const size_t orow = (size_t)(q0 + 8 * half + r) * D_MODEL + head * HEAD_D;
    Ctxb[orow + l16]      = f2bf(o0[r] * inv);
    Ctxb[orow + 16 + l16] = f2bf(o1[r] * inv);
    Ctxb[orow + 32 + l16] = f2bf(o2[r] * inv);
    Ctxb[orow + 48 + l16] = f2bf(o3[r] * inv);
  }
}

// ---------------------------------------------------------------- host-side orchestration
extern "C" void kernel_launch(void* const* d_in, const int* in_sizes, int n_in,
                              void* d_out, int out_size, void* d_ws, size_t ws_size,
                              hipStream_t stream) {
  (void)in_sizes; (void)n_in; (void)out_size; (void)ws_size;
  const float* x  = (const float*)d_in[0];
  const float* Wq = (const float*)d_in[1];
  const float* Wk = (const float*)d_in[2];
  const float* Wv = (const float*)d_in[3];
  const float* Wo = (const float*)d_in[4];
  // d_in[5] = start_pos (always 0 for this harness)

  char* ws = (char*)d_ws;
  size_t off = 0;
  auto alloc = [&](size_t bytes) -> void* {
    void* p = ws + off;
    off += (bytes + 255) & ~(size_t)255;
    return p;
  };

  unsigned short* xb   = (unsigned short*)alloc((size_t)T_SEQ  * D_MODEL * 2);
  unsigned short* Wqb  = (unsigned short*)alloc((size_t)D_MODEL * D_MODEL * 2);
  unsigned short* Wkb  = (unsigned short*)alloc((size_t)KV_DIM  * D_MODEL * 2);
  unsigned short* Wvb  = (unsigned short*)alloc((size_t)KV_DIM  * D_MODEL * 2);
  unsigned short* Wob  = (unsigned short*)alloc((size_t)D_MODEL * D_MODEL * 2);
  float*          Qf   = (float*)alloc((size_t)T_SEQ * D_MODEL * 4);
  float*          Kf   = (float*)alloc((size_t)T_SEQ * KV_DIM  * 4);
  float*          Vf   = (float*)alloc((size_t)T_SEQ * KV_DIM  * 4);
  unsigned short* Qb   = (unsigned short*)alloc((size_t)T_SEQ * D_MODEL * 2);
  unsigned short* Kb   = (unsigned short*)alloc((size_t)T_SEQ * KV_DIM  * 2);
  unsigned short* Vt   = (unsigned short*)alloc((size_t)KV_DIM * T_SEQ  * 2);
  unsigned short* Ctxb = (unsigned short*)alloc((size_t)T_SEQ * D_MODEL * 2);
  float*          cosT = (float*)alloc((size_t)T_SEQ * 32 * 4);
  float*          sinT = (float*)alloc((size_t)T_SEQ * 32 * 4);

  rope_tables_kernel<<<(T_SEQ * 32 + 255) / 256, 256, 0, stream>>>(cosT, sinT);

  cast_f32_bf16_kernel<<<2048, 256, 0, stream>>>(x,  xb,  T_SEQ  * D_MODEL);
  cast_f32_bf16_kernel<<<2048, 256, 0, stream>>>(Wq, Wqb, D_MODEL * D_MODEL);
  cast_f32_bf16_kernel<<<1024, 256, 0, stream>>>(Wk, Wkb, KV_DIM  * D_MODEL);
  cast_f32_bf16_kernel<<<1024, 256, 0, stream>>>(Wv, Wvb, KV_DIM  * D_MODEL);
  cast_f32_bf16_kernel<<<2048, 256, 0, stream>>>(Wo, Wob, D_MODEL * D_MODEL);

  const int tilesQ  = (T_SEQ / 16) * (D_MODEL / 64);   // 4096
  const int tilesKV = (T_SEQ / 16) * (KV_DIM  / 64);   // 1024
  gemm_bf16_nt_kernel<<<(tilesQ  + 7) / 8, 256, 0, stream>>>(xb, Wqb, Qf, T_SEQ, D_MODEL, D_MODEL);
  gemm_bf16_nt_kernel<<<(tilesKV + 7) / 8, 256, 0, stream>>>(xb, Wkb, Kf, T_SEQ, KV_DIM,  D_MODEL);
  gemm_bf16_nt_kernel<<<(tilesKV + 7) / 8, 256, 0, stream>>>(xb, Wvb, Vf, T_SEQ, KV_DIM,  D_MODEL);

  rope_apply_kernel<<<(T_SEQ * N_HEADS * 32 + 255) / 256, 256, 0, stream>>>(Qf, Qb, cosT, sinT, N_HEADS, 0.125f);
  rope_apply_kernel<<<(T_SEQ * N_KVH   * 32 + 255) / 256, 256, 0, stream>>>(Kf, Kb, cosT, sinT, N_KVH, 1.0f);
  transpose_v_kernel<<<(T_SEQ * KV_DIM + 255) / 256, 256, 0, stream>>>(Vf, Vt);

  attention_kernel<<<N_HEADS * (T_SEQ / 64), 128, 0, stream>>>(Qb, Kb, Vt, Ctxb);

  gemm_bf16_nt_kernel<<<(tilesQ + 7) / 8, 256, 0, stream>>>(Ctxb, Wob, (float*)d_out, T_SEQ, D_MODEL, D_MODEL);
}
